// DecomposedQuantize_22582938042421
// MI455X (gfx1250) — matrix operand order repaired and checked by reading, not compiled, not run on hardware
//
#include <hip/hip_runtime.h>

#define Bc 32
#define Lc 8
#define Sc 512
#define Dc 256
#define Ec 2048
#define Nn (Bc * Sc)          // rows per l = 16384
#define DECAYF 0.999f
#define ONE_M_DECAY 0.001f
#define EPSF 1e-5f

typedef __attribute__((ext_vector_type(16))) __bf16 bf16x16;
typedef __attribute__((ext_vector_type(8)))  __bf16 bf16x8;
typedef __attribute__((ext_vector_type(8)))  float  f32x8;

__device__ __forceinline__ unsigned short f2bf(float f) {
    unsigned int u = __builtin_bit_cast(unsigned int, f);
    u += 0x7FFFu + ((u >> 16) & 1u);      // round-to-nearest-even
    return (unsigned short)(u >> 16);
}

// ---------------------------------------------------------------------------
// 1) x (B,L,S,D) f32  ->  xb (L, N=B*S, D) bf16
// ---------------------------------------------------------------------------
__global__ __launch_bounds__(256) void pack_x(const float* __restrict__ x,
                                              unsigned short* __restrict__ xb) {
    size_t idx = (size_t)blockIdx.x * 256 + threadIdx.x;   // flat (b,l,s,d)
    int d = (int)(idx & 255);
    size_t r = idx >> 8;                                   // (b,l,s)
    int s = (int)(r & 511);
    int l = (int)((r >> 9) & 7);
    int b = (int)(r >> 12);
    size_t o = (((size_t)l * Nn) + (size_t)b * Sc + s) * Dc + d;
    xb[o] = f2bf(x[idx]);
}

// ---------------------------------------------------------------------------
// 2) embed (L,D,E) f32 -> embT (L,E,D) bf16 + embT32 (L,E,D) f32
// ---------------------------------------------------------------------------
__global__ __launch_bounds__(256) void trans_embed(const float* __restrict__ embed,
                                                   unsigned short* __restrict__ embT,
                                                   float* __restrict__ embT32) {
    size_t idx = (size_t)blockIdx.x * 256 + threadIdx.x;   // flat (l,e,d)
    int d = (int)(idx & 255);
    int e = (int)((idx >> 8) & 2047);
    int l = (int)(idx >> 19);
    float v = embed[((size_t)l * Dc + d) * Ec + e];
    embT32[idx] = v;
    embT[idx] = f2bf(v);
}

// ---------------------------------------------------------------------------
// 3) ee[l,e] = sum_d embT32[l,e,d]^2
// ---------------------------------------------------------------------------
__global__ __launch_bounds__(256) void ee_kernel(const float* __restrict__ embT32,
                                                 float* __restrict__ ee) {
    int idx = blockIdx.x * 256 + threadIdx.x;              // 0 .. L*E-1
    const float* p = embT32 + (size_t)idx * Dc;
    float s = 0.f;
    #pragma unroll 8
    for (int d = 0; d < Dc; ++d) s += p[d] * p[d];
    ee[idx] = s;
}

// ---------------------------------------------------------------------------
// 4) WMMA argmin: per l, scores = xb_l (N x D) * embT_l^T (D x E);
//    dist = ee - 2*score; track argmin over e.
//    Grid: L*128 blocks; 256 thr = 8 waves; wave -> 16 rows x 32 cols/stage
//    (two independent accumulate chains interleave in the matrix pipe).
// ---------------------------------------------------------------------------
#define LDS_STRIDE 272   // ushorts per row (32B pad -> spreads banks, keeps 32B align)

__global__ __launch_bounds__(256) void argmin_wmma(
    const unsigned short* __restrict__ xb,    // (L, N, D) bf16
    const unsigned short* __restrict__ embT,  // (L, E, D) bf16
    const float* __restrict__ ee,             // (L, E)
    int* __restrict__ idsInt,                 // (B,L,S) flat
    float* __restrict__ idsOut)               // d_out ids region (float)
{
    __shared__ unsigned short ldsB[32 * LDS_STRIDE];   // 32 e-rows x 256 d (+pad)

    const int l        = blockIdx.x >> 7;            // / 128
    const int blockRow = (blockIdx.x & 127) * 128;
    const int tid  = threadIdx.x;
    const int wave = tid >> 5;
    const int lane = tid & 31;
    const int half = lane >> 4;                      // 0 | 1
    const int lcol = lane & 15;
    const int rowBase = blockRow + wave * 16;

    // ---- preload A fragments: 16 rows x 256 K, contiguous k order (k = 16*half + s), same as B
    bf16x16 afrag[8];
    {
        const int arow = rowBase + lcol;             // lane<16: M=lane; lane>=16: M=lane-16
        const unsigned short* xrow = xb + ((size_t)l * Nn + arow) * Dc;
        #pragma unroll
        for (int kc = 0; kc < 8; ++kc) {
            int d0 = kc * 32 + half * 16;            // half0: K{0..15}; half1: K{16..31}
            bf16x8 lo = *(const bf16x8*)(const void*)(xrow + d0);
            bf16x8 hi = *(const bf16x8*)(const void*)(xrow + d0 + 8);
            afrag[kc] = __builtin_shufflevector(lo, hi, 0,1,2,3,4,5,6,7,8,9,10,11,12,13,14,15);
        }
    }

    float bestVal[8];
    int   bestIdx[8];
    #pragma unroll
    for (int r = 0; r < 8; ++r) { bestVal[r] = 3.4e38f; bestIdx[r] = 0; }

    for (int t = 0; t < Ec / 32; ++t) {
        const int e0 = t * 32;
        __syncthreads();                              // protect previous tile
        {
            // stage 32x256 bf16 embed tile: thread -> 64B (two 32B chunks)
            int er = tid >> 3;                        // 0..31 (e local)
            int dc = (tid & 7) * 32;                  // 0..224
            const unsigned short* src = embT + ((size_t)l * Ec + e0 + er) * Dc + dc;
            unsigned short* dst = &ldsB[er * LDS_STRIDE + dc];
            *(bf16x16*)(void*)dst        = *(const bf16x16*)(const void*)src;
            *(bf16x16*)(void*)(dst + 16) = *(const bf16x16*)(const void*)(src + 16);
            if (t + 1 < Ec / 32)                      // hint next tile (global_prefetch_b8)
                __builtin_prefetch(embT + ((size_t)l * Ec + e0 + 32 + er) * Dc + dc, 0, 1);
        }
        __syncthreads();

        f32x8 acc0 = {0.f, 0.f, 0.f, 0.f, 0.f, 0.f, 0.f, 0.f};
        f32x8 acc1 = {0.f, 0.f, 0.f, 0.f, 0.f, 0.f, 0.f, 0.f};
        #pragma unroll
        for (int kc = 0; kc < 8; ++kc) {
            // B layout: lane<16 -> N=lane, K{kc*32..+15}; lane>=16 -> N=lane-16, K{kc*32+16..+31}
            const unsigned short* bp0 = &ldsB[lcol * LDS_STRIDE + kc * 32 + half * 16];
            const unsigned short* bp1 = bp0 + 16 * LDS_STRIDE;
            bf16x16 bfrag0 = *(const bf16x16*)(const void*)bp0;
            bf16x16 bfrag1 = *(const bf16x16*)(const void*)bp1;
            acc0 = __builtin_amdgcn_wmma_f32_16x16x32_bf16(
                false, afrag[kc], false, bfrag0, (short)0, acc0, false, false);
            acc1 = __builtin_amdgcn_wmma_f32_16x16x32_bf16(
                false, afrag[kc], false, bfrag1, (short)0, acc1, false, false);
        }

        const int colA = e0 + lcol;
        const int colB = colA + 16;
        const float eeA = ee[l * Ec + colA];
        const float eeB = ee[l * Ec + colB];
        #pragma unroll
        for (int r = 0; r < 8; ++r) {
            float dA = eeA - 2.0f * acc0[r];          // smaller e first (tie -> first idx)
            if (dA < bestVal[r]) { bestVal[r] = dA; bestIdx[r] = colA; }
            float dB = eeB - 2.0f * acc1[r];
            if (dB < bestVal[r]) { bestVal[r] = dB; bestIdx[r] = colB; }
        }
    }

    // ---- cross-lane argmin within each 16-lane group (first-occurrence ties)
    #pragma unroll
    for (int r = 0; r < 8; ++r) {
        float v = bestVal[r];
        int   i = bestIdx[r];
        #pragma unroll
        for (int m = 1; m < 16; m <<= 1) {
            float v2 = __shfl_xor(v, m, 16);
            int   i2 = __shfl_xor(i, m, 16);
            if (v2 < v || (v2 == v && i2 < i)) { v = v2; i = i2; }
        }
        bestVal[r] = v; bestIdx[r] = i;
    }

    if (lcol == 0) {
        #pragma unroll
        for (int r = 0; r < 8; ++r) {
            int row = rowBase + r + half * 8;         // C/D layout: vgpr r -> M=r (+8 for lanes>=16)
            int b = row >> 9, s = row & 511;
            size_t oi = ((size_t)b * Lc + l) * Sc + s;
            idsInt[oi] = bestIdx[r];
            idsOut[oi] = (float)bestIdx[r];
        }
    }
}

// ---------------------------------------------------------------------------
// 5) gather q -> out, diff reduction, scatter counts/embed_sum
//    one block per (b,l,s) row, 256 threads = D
// ---------------------------------------------------------------------------
__global__ __launch_bounds__(256) void gather_scatter(
    const float* __restrict__ x, const int* __restrict__ idsInt,
    const float* __restrict__ embT32,
    float* __restrict__ out, float* __restrict__ diffAcc,
    float* __restrict__ esum, float* __restrict__ counts)
{
    __shared__ float red[256];
    const int idx = blockIdx.x;                       // flat (b,l,s)
    const int d   = threadIdx.x;
    const int l   = (idx >> 9) & 7;
    const int id  = idsInt[idx];

    float xv = x[(size_t)idx * Dc + d];
    float q  = embT32[((size_t)l * Ec + id) * Dc + d];
    out[(size_t)idx * Dc + d] = q;                    // out = x + sg(q - x) = q

    atomicAdd(&esum[((size_t)l * Ec + id) * Dc + d], xv);
    if (d == 0) atomicAdd(&counts[l * Ec + id], 1.0f);

    if (l == Lc - 1) {
        float dd = q - xv;
        red[d] = dd * dd;
        __syncthreads();
        for (int st = 128; st > 0; st >>= 1) {
            if (d < st) red[d] += red[d + st];
            __syncthreads();
        }
        if (d == 0) {
            // diff = 2/L * mean over (B,S,D):  scale = 2 / (L * B*S*D)
            const float scale = 2.0f / ((float)Lc * (float)Bc * (float)Sc * (float)Dc);
            atomicAdd(diffAcc, red[0] * scale);
        }
    }
}

// ---------------------------------------------------------------------------
// 6) new_cluster_size + per-l sum n
// ---------------------------------------------------------------------------
__global__ __launch_bounds__(256) void ncs_kernel(const float* __restrict__ cs,
                                                  const float* __restrict__ counts,
                                                  float* __restrict__ ncsOut,
                                                  float* __restrict__ nArr) {
    __shared__ float red[256];
    const int l = blockIdx.x, tid = threadIdx.x;
    float local = 0.f;
    for (int j = tid; j < Ec; j += 256) {
        float v = DECAYF * cs[l * Ec + j] + ONE_M_DECAY * counts[l * Ec + j];
        ncsOut[l * Ec + j] = v;
        local += v;
    }
    red[tid] = local;
    __syncthreads();
    for (int st = 128; st > 0; st >>= 1) {
        if (tid < st) red[tid] += red[tid + st];
        __syncthreads();
    }
    if (tid == 0) nArr[l] = red[0];
}

// ---------------------------------------------------------------------------
// 7) new_embed_avg + new_embed over (L,D,E)
// ---------------------------------------------------------------------------
__global__ __launch_bounds__(256) void embed_update(
    const float* __restrict__ ea, const float* __restrict__ esum,
    const float* __restrict__ ncs, const float* __restrict__ nArr,
    float* __restrict__ neaOut, float* __restrict__ newEmbedOut)
{
    size_t idx = (size_t)blockIdx.x * 256 + threadIdx.x;  // flat (l,d,e)
    int e  = (int)(idx & 2047);
    int dd = (int)((idx >> 11) & 255);
    int l  = (int)(idx >> 19);
    float nea = DECAYF * ea[idx] + ONE_M_DECAY * esum[((size_t)l * Ec + e) * Dc + dd];
    neaOut[idx] = nea;
    float n = nArr[l];
    float c = ncs[l * Ec + e];
    float csn = (c + EPSF) / (n + (float)Ec * EPSF) * n;
    newEmbedOut[idx] = nea / csn;
}

// ---------------------------------------------------------------------------
extern "C" void kernel_launch(void* const* d_in, const int* in_sizes, int n_in,
                              void* d_out, int out_size, void* d_ws, size_t ws_size,
                              hipStream_t stream) {
    (void)in_sizes; (void)n_in; (void)out_size; (void)ws_size;
    const float* x     = (const float*)d_in[0];
    const float* embed = (const float*)d_in[1];
    const float* cs    = (const float*)d_in[2];
    const float* ea    = (const float*)d_in[3];
    float* out = (float*)d_out;

    // d_out layout (floats)
    const size_t OUT0 = 0;
    const size_t DIFF = (size_t)Bc * Lc * Sc * Dc;            // 33554432
    const size_t IDS  = DIFF + 1;
    const size_t NEWE = IDS + (size_t)Bc * Lc * Sc;           // new_embed
    const size_t NCS  = NEWE + (size_t)Lc * Dc * Ec;
    const size_t NEA  = NCS + (size_t)Lc * Ec;

    // workspace layout
    char* ws = (char*)d_ws;
    size_t off = 0;
    unsigned short* xb     = (unsigned short*)(ws + off); off += (size_t)Lc * Nn * Dc * 2;
    unsigned short* embT   = (unsigned short*)(ws + off); off += (size_t)Lc * Ec * Dc * 2;
    float*          embT32 = (float*)(ws + off);          off += (size_t)Lc * Ec * Dc * 4;
    float*          eeBuf  = (float*)(ws + off);          off += (size_t)Lc * Ec * 4;
    int*            idsInt = (int*)(ws + off);            off += (size_t)Bc * Lc * Sc * 4;
    float*          esum   = (float*)(ws + off);          off += (size_t)Lc * Ec * Dc * 4;
    float*          counts = (float*)(ws + off);          off += (size_t)Lc * Ec * 4;
    float*          nArr   = (float*)(ws + off);          off += (size_t)Lc * 4;

    hipMemsetAsync(esum,   0, (size_t)Lc * Ec * Dc * 4, stream);
    hipMemsetAsync(counts, 0, (size_t)Lc * Ec * 4, stream);
    hipMemsetAsync(out + DIFF, 0, sizeof(float), stream);

    pack_x       <<<131072, 256, 0, stream>>>(x, xb);
    trans_embed  <<<16384,  256, 0, stream>>>(embed, embT, embT32);
    ee_kernel    <<<64,     256, 0, stream>>>(embT32, eeBuf);
    argmin_wmma  <<<1024,   256, 0, stream>>>(xb, embT, eeBuf, idsInt, out + IDS);
    gather_scatter<<<131072, 256, 0, stream>>>(x, idsInt, embT32,
                                               out + OUT0, out + DIFF, esum, counts);
    ncs_kernel   <<<Lc,     256, 0, stream>>>(cs, counts, out + NCS, nArr);
    embed_update <<<16384,  256, 0, stream>>>(ea, esum, out + NCS, nArr,
                                              out + NEA, out + NEWE);
}
